// LSTModel_29326036697270
// MI455X (gfx1250) — matrix operand order repaired and checked by reading, not compiled, not run on hardware
//
#include <hip/hip_runtime.h>
#include <hip/hip_bf16.h>

// ---------------- problem constants ----------------
#define VOCABN  50000
#define EMBEDN  50
#define NFEAT   3
#define TSTEPS  120
#define HID     300
#define NCLASS  5
#define BATCH   1024
#define DIN     150              // NFEAT*EMBEDN
#define FOURH   1200             // 4*HID
#define KTOT    450              // DIN + HID
#define KPAD    480              // 15 * 32
#define KB      15               // K blocks of 32
#define UPAD    304              // padded hidden units
#define NPAD    1216             // UPAD*4 gate-interleaved columns
#define NT      76               // NPAD/16 column tiles
#define ASTRIDE 488              // LDS A row stride in halfs (bank-friendly)
#define BTILE   16               // batch rows per workgroup
#define NWAVES  16
#define NTHREADS 512
#define NWG     (BATCH / BTILE)  // 64

typedef _Float16 v16h __attribute__((ext_vector_type(16)));
typedef _Float16 v8h  __attribute__((ext_vector_type(8)));
typedef float    v8f  __attribute__((ext_vector_type(8)));

// workspace layout
#define WB_HALFS ((size_t)NT * KB * 512)           // 583,680 halfs
#define BPERM_OFF (WB_HALFS * 2)                   // bytes; 1,167,360 (256-aligned)

#define SMEM_A   (BTILE * ASTRIDE * 2)             // 15,616 B  (f16 A staging)
#define SMEM_C   (BTILE * UPAD * 4)                // 19,456 B  (f32 cell state)
#define SMEM_Z   (BTILE * NPAD * 4)                // 77,824 B  (f32 gate pre-activations)
#define SMEM_Y   (BTILE * 8 * 4)                   //    512 B  (y accumulator)
#define SMEM_U   (HID * NCLASS * 4)                //  6,000 B  (U cached)
#define SMEM_BYTES (SMEM_A + SMEM_C + SMEM_Z + SMEM_Y + SMEM_U)   // 119,408 B

// ---------------- prep: swizzle [Wx;Wh] into WMMA B-fragment blocks ----------------
// Block (j,kb) holds the 32x16 f16 B-tile for column tile j, K-block kb, laid out so
// lane l reads 16 contiguous halfs: n = j*16 + (l&15), K = kb*32 + (l>>4)*16 + e.
// Column permutation: n -> unit u = n>>2, gate g = n&3, original col = g*HID + u.
__global__ void prep_wblocked(const float* __restrict__ Wx, const float* __restrict__ Wh,
                              _Float16* __restrict__ Wb) {
    int blk = blockIdx.x;          // j*KB + kb
    int j   = blk / KB;
    int kb  = blk % KB;
    int tid = threadIdx.x;         // 0..511
    int lane = tid >> 4;
    int e    = tid & 15;
    int n    = j * 16 + (lane & 15);
    int unit = n >> 2;
    int gate = n & 3;
    int k    = kb * 32 + (lane >> 4) * 16 + e;
    float w = 0.f;
    if (unit < HID) {
        int col = gate * HID + unit;
        if (k < DIN)       w = Wx[k * FOURH + col];
        else if (k < KTOT) w = Wh[(k - DIN) * FOURH + col];
    }
    Wb[(size_t)blk * 512 + tid] = (_Float16)w;
}

__global__ void prep_bperm(const float* __restrict__ b, float* __restrict__ bp) {
    int n = blockIdx.x * 608 + threadIdx.x;
    if (n < NPAD) {
        int unit = n >> 2, gate = n & 3;
        bp[n] = (unit < HID) ? b[gate * HID + unit] : 0.f;
    }
}

// ---------------- fused LSTM ----------------
__device__ __forceinline__ float sigm_f(float x) { return 1.f / (1.f + __expf(-x)); }
__device__ __forceinline__ float tanh_f(float x) {
    float e = __expf(2.f * x);
    return 1.f - 2.f / (e + 1.f);
}

__global__ void __launch_bounds__(NTHREADS)
lstm_fused(const int* __restrict__ ids, const float* __restrict__ embed,
           const _Float16* __restrict__ Wb, const float* __restrict__ bperm,
           const float* __restrict__ U, const float* __restrict__ b2,
           float* __restrict__ out) {
    extern __shared__ char smem[];
    _Float16* Ash = (_Float16*)smem;                          // [BTILE][ASTRIDE] f16
    float*    Csh = (float*)(smem + SMEM_A);                  // [BTILE][UPAD]
    float*    Zsh = (float*)(smem + SMEM_A + SMEM_C);         // [BTILE][NPAD]
    float*    Ysh = (float*)(smem + SMEM_A + SMEM_C + SMEM_Z);// [BTILE][8]
    float*    Ush = Ysh + BTILE * 8;                          // [HID][NCLASS]

    const int tid  = threadIdx.x;
    const int b0   = blockIdx.x * BTILE;
    const int lane = tid & 31;
    const int wave = tid >> 5;
    const int lrow  = lane & 15;     // M row / N col within tile
    const int lhalf = lane >> 4;

    // ---- init: cache U, zero cell state, zero h+pad region of A ----
    for (int i = tid; i < HID * NCLASS; i += NTHREADS) Ush[i] = U[i];
    for (int i = tid; i < BTILE * UPAD; i += NTHREADS) Csh[i] = 0.f;
    for (int i = tid; i < BTILE * (ASTRIDE - DIN); i += NTHREADS) {
        int r = i / (ASTRIDE - DIN), c = i % (ASTRIDE - DIN);
        Ash[r * ASTRIDE + DIN + c] = (_Float16)0.f;
    }
    __syncthreads();

    for (int t = 0; t < TSTEPS; ++t) {
        // ---- P1: embedding gather -> x_t (f16), zero y accumulator ----
        for (int i = tid; i < BTILE * DIN; i += NTHREADS) {
            int r = i / DIN, c = i % DIN;
            int f = c / EMBEDN, e = c % EMBEDN;
            int id = ids[((b0 + r) * TSTEPS + t) * NFEAT + f];
            Ash[r * ASTRIDE + c] = (_Float16)embed[id * EMBEDN + e];
        }
        if (tid < BTILE * 8) Ysh[tid] = 0.f;
        __syncthreads();

        // ---- P2: z = [x_t | h] @ Wblocked + b   (WMMA f16 -> f32) ----
        const _Float16* ap = Ash + lrow * ASTRIDE + lhalf * 16;
        for (int j = wave; j < NT; j += NWAVES) {
            float bias = bperm[j * 16 + lrow];
            v8f acc = {bias, bias, bias, bias, bias, bias, bias, bias};
            const _Float16* wp = Wb + (size_t)j * KB * 512 + lane * 16;
#pragma unroll
            for (int kb = 0; kb < KB; ++kb) {
                v8h alo = *(const v8h*)(ap + kb * 32);
                v8h ahi = *(const v8h*)(ap + kb * 32 + 8);
                v16h a = __builtin_shufflevector(alo, ahi,
                          0,1,2,3,4,5,6,7,8,9,10,11,12,13,14,15);
                v16h bm = *(const v16h*)(wp + (size_t)kb * 512);
                acc = __builtin_amdgcn_wmma_f32_16x16x32_f16(
                          false, a, false, bm, (short)0, acc, false, false);
            }
            int col   = j * 16 + lrow;
            int rbase = lhalf * 8;
#pragma unroll
            for (int v = 0; v < 8; ++v)
                Zsh[(rbase + v) * NPAD + col] = acc[v];
        }
        __syncthreads();

        // ---- P3: gates, cell/h update, partial y = h @ U ----
        {
            int r  = tid >> 5;            // 0..15 batch row
            int u0 = (tid & 31) * 10;     // 32 threads/row * 10 units = 320 >= 304
            float yacc[NCLASS] = {0.f, 0.f, 0.f, 0.f, 0.f};
            for (int u = u0; u < u0 + 10 && u < UPAD; ++u) {
                float4 z = *(const float4*)(Zsh + r * NPAD + 4 * u);
                float ig = sigm_f(z.x);
                float fg = sigm_f(z.y);
                float gg = tanh_f(z.z);
                float og = sigm_f(z.w);
                float c  = fg * Csh[r * UPAD + u] + ig * gg;
                Csh[r * UPAD + u] = c;
                float h = og * tanh_f(c);
                Ash[r * ASTRIDE + DIN + u] = (_Float16)h;   // h_t for next step's GEMM
                if (u < HID) {
#pragma unroll
                    for (int cl = 0; cl < NCLASS; ++cl)
                        yacc[cl] += h * Ush[u * NCLASS + cl];
                }
            }
#pragma unroll
            for (int cl = 0; cl < NCLASS; ++cl)
                atomicAdd(&Ysh[r * 8 + cl], yacc[cl]);
        }
        __syncthreads();

        // ---- P4: emit y_t ----
        if (tid < BTILE * NCLASS) {
            int r = tid / NCLASS, cl = tid % NCLASS;
            out[((size_t)(b0 + r) * TSTEPS + t) * NCLASS + cl] = Ysh[r * 8 + cl] + b2[cl];
        }
        __syncthreads();
    }
}

// ---------------- launcher ----------------
extern "C" void kernel_launch(void* const* d_in, const int* in_sizes, int n_in,
                              void* d_out, int out_size, void* d_ws, size_t ws_size,
                              hipStream_t stream) {
    (void)in_sizes; (void)n_in; (void)out_size; (void)ws_size;
    const int*   ids   = (const int*)d_in[0];
    const float* embed = (const float*)d_in[1];
    const float* Wx    = (const float*)d_in[2];
    const float* Wh    = (const float*)d_in[3];
    const float* b     = (const float*)d_in[4];
    const float* U     = (const float*)d_in[5];
    const float* b2    = (const float*)d_in[6];
    float*       out   = (float*)d_out;

    _Float16* Wb    = (_Float16*)d_ws;
    float*    bperm = (float*)((char*)d_ws + BPERM_OFF);

    // allow >64KB dynamic LDS (CDNA5 WGP has 320KB)
    (void)hipFuncSetAttribute((const void*)lstm_fused,
                              hipFuncAttributeMaxDynamicSharedMemorySize, SMEM_BYTES);

    prep_wblocked<<<NT * KB, 512, 0, stream>>>(Wx, Wh, Wb);
    prep_bperm<<<2, 608, 0, stream>>>(b, bperm);
    lstm_fused<<<NWG, NTHREADS, SMEM_BYTES, stream>>>(ids, embed, Wb, bperm, U, b2, out);
}
